// NeuralSamplerPosEmbLearnableLargeEnergyNN_77747497992229
// MI455X (gfx1250) — matrix operand order repaired and checked by reading, not compiled, not run on hardware
//
#include <hip/hip_runtime.h>
#include <math.h>

// ---------------------------------------------------------------------------
// Problem constants
// ---------------------------------------------------------------------------
#define BB   64      // batch
#define TT   1024    // sequence length
#define DIN  128     // input feature dim
#define DPRE 256     // pre_linear width / LSTM input
#define HH   128     // LSTM hidden per direction (2*LATENT)
#define GG   512     // 4*HH gates
#define LL   512     // OUT_LEN
#define MROWS (BB*TT) // 65536

// ---------------------------------------------------------------------------
// WMMA bf16 helpers (gfx1250, wave32)
// ---------------------------------------------------------------------------
typedef __attribute__((ext_vector_type(16))) __bf16 bf16x16;
typedef __attribute__((ext_vector_type(8)))  __bf16 bf16x8;
typedef __attribute__((ext_vector_type(8)))  float  f32x8;

union frag_u { bf16x16 v; bf16x8 h[2]; };

// A matrix 16x32 bf16 (MxK): row m = lane%16.
// lane<16 : elems = K{kb+0..7},  K{kb+16..23}
// lane>=16: elems = K{kb+8..15}, K{kb+24..31}
__device__ __forceinline__ bf16x16 frag_a_load(const __bf16* base, int ld,
                                               int row0, int kb, int lane) {
  int m = row0 + (lane & 15);
  const __bf16* p = base + (size_t)m * ld + kb + ((lane >> 4) << 3);
  frag_u u;
  u.h[0] = *(const bf16x8*)p;
  u.h[1] = *(const bf16x8*)(p + 16);
  return u.v;
}

// B matrix 32x16 bf16 (KxN) taken from row-major W (N,K): col n = lane%16.
// lanes 0-15 hold K = kb+0..7, kb+16..23; lanes 16-31 hold K = kb+8..15, kb+24..31.
__device__ __forceinline__ bf16x16 frag_b_load(const __bf16* base, int ld,
                                               int n0, int kb, int lane) {
  int n = n0 + (lane & 15);
  const __bf16* p = base + (size_t)n * ld + kb + ((lane >> 4) << 3);
  frag_u u;
  u.h[0] = *(const bf16x8*)p;
  u.h[1] = *(const bf16x8*)(p + 16);
  return u.v;
}

__device__ __forceinline__ f32x8 wmma_bf16(bf16x16 a, bf16x16 b, f32x8 c) {
  return __builtin_amdgcn_wmma_f32_16x16x32_bf16(false, a, false, b,
                                                 (short)0, c, false, false);
}

__device__ __forceinline__ float sigf(float x) { return 1.0f / (1.0f + expf(-x)); }

// Fast gate nonlinearities for the recurrent loop: native v_exp_f32 + v_rcp_f32
// (error ~1e-6 rel, far below the bf16 quantization already in this path).
__device__ __forceinline__ float fast_sig(float x) {
  return __builtin_amdgcn_rcpf(1.0f + __expf(-x));
}
__device__ __forceinline__ float fast_tanh(float x) {
  return 2.0f * fast_sig(2.0f * x) - 1.0f;
}

// ---------------------------------------------------------------------------
// Block reductions (wave32)
// ---------------------------------------------------------------------------
__device__ __forceinline__ float blockReduceSum(float v, float* red) {
  int lane = threadIdx.x & 31, w = threadIdx.x >> 5, nw = blockDim.x >> 5;
#pragma unroll
  for (int o = 16; o; o >>= 1) v += __shfl_xor(v, o, 32);
  if (!lane) red[w] = v;
  __syncthreads();
  float r = (w == 0 && lane < nw) ? red[lane] : 0.0f;
  if (w == 0) {
#pragma unroll
    for (int o = 16; o; o >>= 1) r += __shfl_xor(r, o, 32);
    if (!lane) red[0] = r;
  }
  __syncthreads();
  r = red[0];
  __syncthreads();
  return r;
}

__device__ __forceinline__ float blockReduceMax(float v, float* red) {
  int lane = threadIdx.x & 31, w = threadIdx.x >> 5, nw = blockDim.x >> 5;
#pragma unroll
  for (int o = 16; o; o >>= 1) v = fmaxf(v, __shfl_xor(v, o, 32));
  if (!lane) red[w] = v;
  __syncthreads();
  float r = (w == 0 && lane < nw) ? red[lane] : -3.4e38f;
  if (w == 0) {
#pragma unroll
    for (int o = 16; o; o >>= 1) r = fmaxf(r, __shfl_xor(r, o, 32));
    if (!lane) red[0] = r;
  }
  __syncthreads();
  r = red[0];
  __syncthreads();
  return r;
}

// ---------------------------------------------------------------------------
// Elementwise utility kernels
// ---------------------------------------------------------------------------
__global__ void f2bf_kernel(const float* __restrict__ src, __bf16* __restrict__ dst, int n) {
  int i = blockIdx.x * blockDim.x + threadIdx.x;
  if (i < n) dst[i] = (__bf16)src[i];
}

__global__ void addv_kernel(const float* __restrict__ a, const float* __restrict__ b,
                            float* __restrict__ o, int n) {
  int i = blockIdx.x * blockDim.x + threadIdx.x;
  if (i < n) o[i] = a[i] + b[i];
}

// mag[b*T+t] = sum_d exp(x[b,t,d]); one wave per row
__global__ void magnitude_kernel(const float* __restrict__ x, float* __restrict__ mag) {
  int gw = blockIdx.x * (blockDim.x >> 5) + (threadIdx.x >> 5);
  int lane = threadIdx.x & 31;
  const float* p = x + (size_t)gw * DIN;
  float a = expf(p[lane]) + expf(p[lane + 32]) + expf(p[lane + 64]) + expf(p[lane + 96]);
#pragma unroll
  for (int o = 16; o; o >>= 1) a += __shfl_xor(a, o, 32);
  if (!lane) mag[gw] = a;
}

// ---------------------------------------------------------------------------
// Generic WMMA GEMM: out = act(A_bf16(M,K) @ W_bf16(N,K)^T + bias)
// Each wave computes a 16x64 output tile. K is a compile-time constant so the
// k-loop fully unrolls: every fragment is a fresh SSA value (no ping-pong
// register copies), addresses fold to immediate offsets, and the scheduler
// pipelines loads across iterations against back-to-back WMMAs.
// Optional (b*T+t)->(t*B+b) output row remap.
// ---------------------------------------------------------------------------
template <int K>
__global__ __launch_bounds__(256)
void gemm_bf16_kernel(const __bf16* __restrict__ A, const __bf16* __restrict__ W,
                      const float* __restrict__ bias, __bf16* __restrict__ outB,
                      float* __restrict__ outF, int M, int N,
                      int relu, int remap_mod, int remap_mul) {
  int gw   = blockIdx.x * (blockDim.x >> 5) + (threadIdx.x >> 5);
  int lane = threadIdx.x & 31;
  int tilesN = N >> 6;
  int total  = (M >> 4) * tilesN;
  if (gw >= total) return;
  int tm = gw / tilesN, tn = gw % tilesN;
  int nb = tn << 6;
  const __bf16* Arow = A + (size_t)(tm << 4) * K;
  const __bf16* Wrow = W + (size_t)nb * K;

  f32x8 c[4];
#pragma unroll
  for (int j = 0; j < 4; ++j)
#pragma unroll
    for (int r = 0; r < 8; ++r) c[j][r] = 0.0f;

#pragma unroll
  for (int kb = 0; kb < K; kb += 32) {
    bf16x16 a = frag_a_load(Arow, K, 0, kb, lane);
#pragma unroll
    for (int j = 0; j < 4; ++j) {
      bf16x16 b = frag_b_load(Wrow, K, j << 4, kb, lane);
      c[j] = wmma_bf16(a, b, c[j]);
    }
  }

  // epilogue: bias + activation + store (remap divide hoisted per-row)
  int nloc = lane & 15;
  float bv[4];
#pragma unroll
  for (int j = 0; j < 4; ++j) bv[j] = bias ? bias[nb + (j << 4) + nloc] : 0.0f;
#pragma unroll
  for (int r = 0; r < 8; ++r) {
    int m = (tm << 4) + r + ((lane >> 4) << 3);
    size_t orow = (size_t)m;
    if (remap_mod) orow = (size_t)(m % remap_mod) * remap_mul + (m / remap_mod);
    size_t rb = orow * N;
#pragma unroll
    for (int j = 0; j < 4; ++j) {
      float v = c[j][r] + bv[j];
      if (relu) v = fmaxf(v, 0.0f);
      int n = nb + (j << 4) + nloc;
      if (outB) outB[rb + n] = (__bf16)v;
      if (outF) outF[rb + n] = v;
    }
  }
}

// ---------------------------------------------------------------------------
// LSTM recurrence for one layer, both directions.
// grid = (2 dirs, 4 batch-groups of 16 rows), 512 threads (16 waves).
// gates_x: (2, T, B, 512) bf16 already holds x@Wih^T + bih + bhh.
// Whh fragments are loop-invariant: preloaded into registers once
// (2 tiles x 4 ksteps x 16 bf16 = 64 VGPRs/wave). All four A fragments are
// loaded from LDS before the WMMA sequence so the 8 WMMAs issue back-to-back
// behind a single dscnt wait cascade. Gate nonlinearity uses native
// v_exp_f32/v_rcp_f32. hout: (T, B, 2, 128) bf16.
// ---------------------------------------------------------------------------
__global__ __launch_bounds__(512)
void lstm_rec_kernel(const __bf16* __restrict__ gatesx, const __bf16* __restrict__ WhhB,
                     __bf16* __restrict__ hout, int layer) {
  __shared__ float gsh[16 * GG];                 // 32 KB gate staging
  __shared__ __align__(16) __bf16 hsh[16 * HH];  // 4 KB hidden state

  const int d  = blockIdx.x;
  const int bg = blockIdx.y;
  const __bf16* gx = gatesx + (size_t)d * TT * BB * GG;
  const __bf16* Wd = WhhB + (size_t)(layer * 2 + d) * GG * HH;

  const int tid = threadIdx.x, lane = tid & 31, w = tid >> 5;
  const int tn0 = w << 1;  // 2 N-tiles per wave, 16 waves -> 32 tiles = 16x512

  // preload recurrent weights into registers (invariant over time)
  bf16x16 wb[2][4];
#pragma unroll
  for (int j = 0; j < 2; ++j)
#pragma unroll
    for (int kk = 0; kk < 4; ++kk)
      wb[j][kk] = frag_b_load(Wd, HH, (tn0 + j) << 4, kk << 5, lane);

  float creg[4] = {0.0f, 0.0f, 0.0f, 0.0f};
  for (int i = tid; i < 16 * HH; i += 512) hsh[i] = (__bf16)0.0f;
  __syncthreads();

  for (int step = 0; step < TT; ++step) {
    const int t = d ? (TT - 1 - step) : step;
    const __bf16* gxt = gx + ((size_t)t * BB + bg * 16) * GG;

    // prefetch next step's 16KB gate tile into cache (global_prefetch_b8)
    if (step + 1 < TT && tid < 128) {
      const int tnext = d ? (TT - 2 - step) : (step + 1);
      const __bf16* gp = gx + ((size_t)tnext * BB + bg * 16) * GG + (size_t)tid * 64;
      __builtin_prefetch((const void*)gp, 0, 3);
    }

    f32x8 c[2];
#pragma unroll
    for (int j = 0; j < 2; ++j) {
      int n = ((tn0 + j) << 4) + (lane & 15);
#pragma unroll
      for (int r = 0; r < 8; ++r) {
        int m = r + ((lane >> 4) << 3);
        c[j][r] = (float)gxt[(size_t)m * GG + n];
      }
    }
    // hoist all A fragments, then 8 back-to-back WMMAs
    bf16x16 afr[4];
#pragma unroll
    for (int kk = 0; kk < 4; ++kk) afr[kk] = frag_a_load(hsh, HH, 0, kk << 5, lane);
#pragma unroll
    for (int kk = 0; kk < 4; ++kk) {
      c[0] = wmma_bf16(afr[kk], wb[0][kk], c[0]);
      c[1] = wmma_bf16(afr[kk], wb[1][kk], c[1]);
    }
#pragma unroll
    for (int j = 0; j < 2; ++j) {
      int n = ((tn0 + j) << 4) + (lane & 15);
#pragma unroll
      for (int r = 0; r < 8; ++r) {
        int m = r + ((lane >> 4) << 3);
        gsh[m * GG + n] = c[j][r];
      }
    }
    __syncthreads();

#pragma unroll
    for (int i = 0; i < 4; ++i) {
      int e = (tid << 2) + i;         // 16*128 = 2048 elements
      int m = e >> 7, hid = e & 127;
      float ig = gsh[m * GG + hid];
      float fg = gsh[m * GG + 128 + hid];
      float gg = gsh[m * GG + 256 + hid];
      float og = gsh[m * GG + 384 + hid];
      float cc = fast_sig(fg) * creg[i] + fast_sig(ig) * fast_tanh(gg);
      creg[i] = cc;
      float hv = fast_sig(og) * fast_tanh(cc);
      __bf16 hb = (__bf16)hv;
      hsh[e] = hb;
      hout[((size_t)t * BB + bg * 16 + m) * (2 * HH) + d * HH + hid] = hb;
    }
    __syncthreads();
  }
}

// ---------------------------------------------------------------------------
// score = sigmoid(hs(row t*B+b) . Ws2 + bs2); one wave per (b,t); out [b*T+t]
// ---------------------------------------------------------------------------
__global__ void score_kernel(const __bf16* __restrict__ hs, const float* __restrict__ Ws2,
                             const float* __restrict__ bs2, float* __restrict__ score) {
  int gw = blockIdx.x * (blockDim.x >> 5) + (threadIdx.x >> 5);
  int lane = threadIdx.x & 31;
  int b = gw >> 10, t = gw & 1023;
  const __bf16* p = hs + (size_t)(t * BB + b) * HH;
  float a = 0.0f;
#pragma unroll
  for (int i = 0; i < 4; ++i) {
    int k = lane + (i << 5);
    a += (float)p[k] * Ws2[k];
  }
#pragma unroll
  for (int o = 16; o; o >>= 1) a += __shfl_xor(a, o, 32);
  if (!lane) score[gw] = sigf(a + bs2[0]);
}

// ---------------------------------------------------------------------------
// _score_norm: one block (1024 threads) per batch row.
// Scale-invariant in its input, so it also serves energy_n directly from mag.
// ---------------------------------------------------------------------------
__global__ __launch_bounds__(1024)
void score_norm_kernel(const float* __restrict__ in, float* __restrict__ outp) {
  __shared__ float red[32];
  int b = blockIdx.x, t = threadIdx.x;
  float v = in[(size_t)b * TT + t];
  float total = blockReduceSum(v, red);
  float s = v / total * (float)LL;
  float mx = blockReduceMax(s, red);
  if (mx >= 1.0f) s = s / mx;
  float sum2 = blockReduceSum(s, red);
  float dist = (float)LL - sum2;
  float iv = (s < 0.99f && s > 0.01f) ? (1.0f - s) : 0.0f;
  float den = blockReduceSum(iv, red);
  float alpha = fminf(dist / fmaxf(den, 1e-12f), 1.0f);
  if (dist >= 1.0f) s = s + iv * alpha;
  outp[(size_t)b * TT + t] = s;
}

// ---------------------------------------------------------------------------
// Selection: per-b cumsum scan, then per (l, d-half) binary search of the
// t-run with cum in (l, l+1], normalized weighted sums of x and pos_emb.
// Also emits per-b std (ddof=1) of s.
// ---------------------------------------------------------------------------
__global__ __launch_bounds__(1024)
void select_kernel(const float* __restrict__ s_in, const float* __restrict__ x,
                   const float* __restrict__ pe, float* __restrict__ feature,
                   float* __restrict__ emb, float* __restrict__ stdv) {
  __shared__ float ss[TT], ca[TT], cb[TT], red[32];
  int b = blockIdx.x, t = threadIdx.x;
  float v = s_in[(size_t)b * TT + t];
  ss[t] = v;
  ca[t] = v;
  float sum = blockReduceSum(v, red);
  float sumsq = blockReduceSum(v * v, red);
  if (t == 0) {
    float mean = sum / (float)TT;
    float var = (sumsq - (float)TT * mean * mean) / (float)(TT - 1);
    stdv[b] = sqrtf(fmaxf(var, 0.0f));
  }
  __syncthreads();

  float* src = ca;
  float* dst = cb;
  for (int off = 1; off < TT; off <<= 1) {
    float val = src[t] + ((t >= off) ? src[t - off] : 0.0f);
    dst[t] = val;
    __syncthreads();
    float* tmp = src; src = dst; dst = tmp;
  }
  // inclusive cumsum now in src[]

  int l = t >> 1, dh = t & 1;
  float lv = (float)l;
  int lo = 0, hi = TT;
  while (lo < hi) { int mid = (lo + hi) >> 1; if (src[mid] > lv) hi = mid; else lo = mid + 1; }
  int t0 = lo;
  float lv1 = lv + 1.0f;
  lo = 0; hi = TT;
  while (lo < hi) { int mid = (lo + hi) >> 1; if (src[mid] > lv1) hi = mid; else lo = mid + 1; }
  int t1 = lo;

  float denom = 1e-8f;
  for (int tt = t0; tt < t1; ++tt) denom += ss[tt];

  for (int dd = dh * 64; dd < dh * 64 + 64; ++dd) {
    float af = 0.0f, ae = 0.0f;
    for (int tt = t0; tt < t1; ++tt) {
      float wn = ss[tt] / denom;
      af += wn * x[((size_t)b * TT + tt) * DIN + dd];
      ae += wn * pe[(size_t)tt * DIN + dd];
    }
    feature[(((size_t)b * 2 + 0) * LL + l) * DIN + dd] = af;
    feature[(((size_t)b * 2 + 1) * LL + l) * DIN + dd] = ae;
    emb[((size_t)b * LL + l) * DIN + dd] = ae;
  }
}

__global__ void loss_kernel(const float* __restrict__ stdv, float* __restrict__ loss) {
  __shared__ float sh[64];
  sh[threadIdx.x] = stdv[threadIdx.x];
  __syncthreads();
  if (threadIdx.x == 0) {
    float a = 0.0f;
    for (int i = 0; i < 64; ++i) a += sh[i];
    *loss = a / 64.0f;
  }
}

// ---------------------------------------------------------------------------
// Host launch
// ---------------------------------------------------------------------------
extern "C" void kernel_launch(void* const* d_in, const int* in_sizes, int n_in,
                              void* d_out, int out_size, void* d_ws, size_t ws_size,
                              hipStream_t stream) {
  (void)in_sizes; (void)n_in; (void)out_size; (void)ws_size;
  const float* x   = (const float*)d_in[0];
  const float* W1  = (const float*)d_in[1];
  const float* b1  = (const float*)d_in[2];
  const float* W2  = (const float*)d_in[3];
  const float* b2  = (const float*)d_in[4];
  const float* Wih = (const float*)d_in[5];
  const float* Whh = (const float*)d_in[6];
  const float* bih = (const float*)d_in[7];
  const float* bhh = (const float*)d_in[8];
  const float* Ws1 = (const float*)d_in[9];
  const float* bs1 = (const float*)d_in[10];
  const float* Ws2 = (const float*)d_in[11];
  const float* bs2 = (const float*)d_in[12];
  const float* pe  = (const float*)d_in[13];

  char* wsb = (char*)d_ws;
  size_t off = 0;
  auto alloc = [&](size_t bytes) -> char* {
    char* p = wsb + off;
    off += (bytes + 255) & ~(size_t)255;
    return p;
  };

  __bf16* xbf  = (__bf16*)alloc((size_t)MROWS * DIN * 2);
  __bf16* W1b  = (__bf16*)alloc((size_t)DPRE * DIN * 2);
  __bf16* W2b  = (__bf16*)alloc((size_t)DPRE * DPRE * 2);
  __bf16* Wihb = (__bf16*)alloc((size_t)4 * GG * DPRE * 2);
  __bf16* Whhb = (__bf16*)alloc((size_t)4 * GG * HH * 2);
  __bf16* Ws1b = (__bf16*)alloc((size_t)HH * DPRE * 2);
  float*  bg   = (float*) alloc((size_t)4 * GG * 4);
  __bf16* pre1 = (__bf16*)alloc((size_t)MROWS * DPRE * 2);
  __bf16* x0   = (__bf16*)alloc((size_t)MROWS * DPRE * 2);   // (T,B,256)
  __bf16* gts  = (__bf16*)alloc((size_t)2 * MROWS * GG * 2); // (2,T,B,512)
  __bf16* h0   = (__bf16*)alloc((size_t)MROWS * 2 * HH * 2); // (T,B,2,128)
  __bf16* h1   = (__bf16*)alloc((size_t)MROWS * 2 * HH * 2);
  __bf16* hs   = (__bf16*)alloc((size_t)MROWS * HH * 2);
  float*  mag  = (float*) alloc((size_t)MROWS * 4);
  float*  scr  = (float*) alloc((size_t)MROWS * 4);
  float*  stdv = (float*) alloc((size_t)64 * 4);

  float* out      = (float*)d_out;
  float* feature  = out;                                   // (B,2,L,128)
  float* emb      = out + (size_t)BB * 2 * LL * DIN;       // (B,L,128)
  float* s_out    = emb + (size_t)BB * LL * DIN;           // (B,T)
  float* loss     = s_out + (size_t)BB * TT;               // scalar
  float* energy_n = loss + 1;                              // (B,T)

  auto f2bf = [&](const float* s, __bf16* d, int n) {
    f2bf_kernel<<<(n + 255) / 256, 256, 0, stream>>>(s, d, n);
  };
  auto gemm = [&](const __bf16* A, const __bf16* W, const float* bias,
                  __bf16* oB, float* oF, int M, int N, int K,
                  int relu, int rmod, int rmul) {
    int waves = (M >> 4) * (N >> 6);
    int blocks = (waves + 7) / 8;
    if (K == 128)
      gemm_bf16_kernel<128><<<blocks, 256, 0, stream>>>(A, W, bias, oB, oF,
                                                        M, N, relu, rmod, rmul);
    else
      gemm_bf16_kernel<256><<<blocks, 256, 0, stream>>>(A, W, bias, oB, oF,
                                                        M, N, relu, rmod, rmul);
  };

  // --- precision conversions + bias fold ---
  f2bf(x,   xbf,  MROWS * DIN);
  f2bf(W1,  W1b,  DPRE * DIN);
  f2bf(W2,  W2b,  DPRE * DPRE);
  f2bf(Wih, Wihb, 4 * GG * DPRE);
  f2bf(Whh, Whhb, 4 * GG * HH);
  f2bf(Ws1, Ws1b, HH * DPRE);
  addv_kernel<<<(4 * GG + 255) / 256, 256, 0, stream>>>(bih, bhh, bg, 4 * GG);

  // --- energy magnitudes ---
  magnitude_kernel<<<MROWS / 8, 256, 0, stream>>>(x, mag);

  // --- pre_linear MLP (WMMA), second GEMM writes (T,B,256) layout ---
  gemm(xbf,  W1b, b1, pre1, nullptr, MROWS, DPRE, DIN,  1, 0,  0);
  gemm(pre1, W2b, b2, x0,   nullptr, MROWS, DPRE, DPRE, 1, TT, BB);

  // --- 2-layer BiLSTM: hoisted x-GEMMs + sequential recurrence ---
  for (int l = 0; l < 2; ++l) {
    const __bf16* Ain = l ? h0 : x0;
    for (int d = 0; d < 2; ++d) {
      gemm(Ain, Wihb + (size_t)(l * 2 + d) * GG * DPRE, bg + (l * 2 + d) * GG,
           gts + (size_t)d * MROWS * GG, nullptr, MROWS, GG, DPRE, 0, 0, 0);
    }
    lstm_rec_kernel<<<dim3(2, 4), 512, 0, stream>>>(gts, Whhb, l ? h1 : h0, l);
  }

  // --- score head ---
  gemm(h1, Ws1b, bs1, hs, nullptr, MROWS, HH, DPRE, 1, 0, 0);
  score_kernel<<<MROWS / 8, 256, 0, stream>>>(hs, Ws2, bs2, scr);

  // --- score_norm for score and energy (scale-invariant: mag works directly) ---
  score_norm_kernel<<<BB, 1024, 0, stream>>>(scr, s_out);
  score_norm_kernel<<<BB, 1024, 0, stream>>>(mag, energy_n);

  // --- selection / outputs / loss ---
  select_kernel<<<BB, 1024, 0, stream>>>(s_out, x, pe, feature, emb, stdv);
  loss_kernel<<<1, 64, 0, stream>>>(stdv, loss);
}
